// TensorMOE_87376814670014
// MI455X (gfx1250) — matrix operand order repaired and checked by reading, not compiled, not run on hardware
//
#include <hip/hip_runtime.h>

// ---------------------------------------------------------------------------
// TensorMOE fused implementation for gfx1250 (MI455X), bf16 WMMA everywhere.
// B=8192, IN_C=OUT_C=1024, HM=HN=64, N_HIDDEN=4 (3 hidden iterations).
// ---------------------------------------------------------------------------

#define BT    8192
#define INC   1024
#define OUTC  1024
#define HD    64          // HM == HN == 64

typedef __attribute__((ext_vector_type(16))) __bf16 v16bf;
typedef __attribute__((ext_vector_type(8)))  __bf16 v8bf;
typedef __attribute__((ext_vector_type(8)))  float  v8f;

// ---- bf16 helpers (RNE) ----------------------------------------------------
static __device__ __forceinline__ __bf16 f2bf(float f) {
  unsigned u = __builtin_bit_cast(unsigned, f);
  unsigned r = u + 0x7FFFu + ((u >> 16) & 1u);
  unsigned short h = (unsigned short)(r >> 16);
  return __builtin_bit_cast(__bf16, h);
}
static __device__ __forceinline__ unsigned f2bf2(float a, float b) {
  unsigned ua = __builtin_bit_cast(unsigned, a);
  unsigned ub = __builtin_bit_cast(unsigned, b);
  unsigned ra = (ua + 0x7FFFu + ((ua >> 16) & 1u)) >> 16;
  unsigned rb = (ub + 0x7FFFu + ((ub >> 16) & 1u)) & 0xFFFF0000u;
  return ra | rb;
}
static __device__ __forceinline__ float bf2f(__bf16 b) {
  unsigned short h = __builtin_bit_cast(unsigned short, b);
  unsigned u = ((unsigned)h) << 16;
  return __builtin_bit_cast(float, u);
}

// ---- WMMA wrapper ----------------------------------------------------------
static __device__ __forceinline__ v8f wmma_bf16(v16bf a, v16bf b, v8f c) {
  return __builtin_amdgcn_wmma_f32_16x16x32_bf16(
      /*neg_a=*/false, a, /*neg_b=*/false, b,
      /*c_mod=*/(short)0, c, /*reuse_a=*/false, /*reuse_b=*/false);
}

// A-fragment 16x32 bf16 from row-major source, row stride lda (halves).
// Lane l: row = l&15; lanes<16 hold K in {k0..k0+7, k0+16..k0+23},
// lanes>=16 hold K in {k0+8..k0+15, k0+24..k0+31}. (ISA 7.12.2)
static __device__ __forceinline__ v16bf load_a_frag(const __bf16* a, int lda,
                                                    int k0, int lane) {
  int r  = lane & 15;
  int kb = k0 + ((lane & 16) ? 8 : 0);
  const __bf16* p = a + (size_t)r * lda + kb;
  v8bf lo = *(const v8bf*)(p);
  v8bf hi = *(const v8bf*)(p + 16);
  v16bf v;
#pragma unroll
  for (int j = 0; j < 8; ++j) { v[j] = lo[j]; v[j + 8] = hi[j]; }
  return v;
}

// A-fragment with gathered K (element (r,k) at a[r*rs + k*es]) — used for the
// "contract over n, per m" hidden step which reads strided LDS.
static __device__ __forceinline__ v16bf load_a_frag_gather(const __bf16* a,
                                                           int rs, int es,
                                                           int k0, int lane) {
  int r  = lane & 15;
  int kb = k0 + ((lane & 16) ? 8 : 0);
  const __bf16* p = a + (size_t)r * rs;
  v16bf v;
#pragma unroll
  for (int j = 0; j < 8; ++j) {
    v[j]     = p[(size_t)(kb + j) * es];
    v[j + 8] = p[(size_t)(kb + 16 + j) * es];
  }
  return v;
}

// B-fragment 32x16 bf16 from K-contiguous (transposed) weight copy:
// element (k, c) at wt[c*ldk + k]. Lanes<16 hold K 0..7,16..23; lanes>=16 K 8..15,24..31.
static __device__ __forceinline__ v16bf load_b_frag(const __bf16* wt, int ldk,
                                                    int k0, int lane) {
  int c  = lane & 15;
  int kb = k0 + ((lane & 16) ? 8 : 0);
  const __bf16* p = wt + (size_t)c * ldk + kb;
  v8bf lo = *(const v8bf*)(p);
  v8bf hi = *(const v8bf*)(p + 16);
  v16bf v;
#pragma unroll
  for (int j = 0; j < 8; ++j) { v[j] = lo[j]; v[j + 8] = hi[j]; }
  return v;
}

static __device__ __forceinline__ int cfrag_col(int lane) { return lane & 15; }
static __device__ __forceinline__ int cfrag_row(int lane, int v) {
  return v + ((lane & 16) ? 8 : 0);
}

// ---------------------------------------------------------------------------
// K0a: fp32 -> bf16 plain copy
// ---------------------------------------------------------------------------
__global__ __launch_bounds__(256) void k_copy_bf16(__bf16* __restrict__ dst,
                                                   const float* __restrict__ src,
                                                   int n) {
  int i = blockIdx.x * 256 + threadIdx.x;
  if (i < n) dst[i] = f2bf(src[i]);
}

// ---------------------------------------------------------------------------
// K0b: tiled transpose + convert: src [D][R][C] f32 -> dst [D][C][R] bf16
// grid.x = D * (R/64) * (C/64)
// ---------------------------------------------------------------------------
__global__ __launch_bounds__(256) void k_transpose_bf16(
    __bf16* __restrict__ dst, const float* __restrict__ src, int R, int C) {
  __shared__ float tile[64][65];
  int tilesR = R >> 6, tilesC = C >> 6;
  int bid = blockIdx.x;
  int d   = bid / (tilesR * tilesC);
  int rm  = bid % (tilesR * tilesC);
  int rt  = rm / tilesC, ct = rm % tilesC;
  const float* s = src + (size_t)d * R * C + (size_t)(rt * 64) * C + ct * 64;
  for (int e = threadIdx.x; e < 4096; e += 256) {
    int rr = e >> 6, cc = e & 63;
    tile[rr][cc] = s[(size_t)rr * C + cc];
  }
  __syncthreads();
  __bf16* o = dst + (size_t)d * R * C + (size_t)(ct * 64) * R + rt * 64;
  for (int e = threadIdx.x; e < 4096; e += 256) {
    int cc = e >> 6, rr = e & 63;
    o[(size_t)cc * R + rr] = f2bf(tile[rr][cc]);
  }
}

// ---------------------------------------------------------------------------
// K2: fused gates + input projections + hidden loop. One block = 16 batch rows.
// LDS: xs 32KB + hbuf 128KB + lbuf 8KB = 168KB (WGP has 320KB).
// ---------------------------------------------------------------------------
__global__ __launch_bounds__(256) void k_hidden(
    const float* __restrict__ x,
    const __bf16* __restrict__ wmbf, const float* __restrict__ wm_b,
    const __bf16* __restrict__ wnbf, const float* __restrict__ wn_b,
    const __bf16* __restrict__ WimT, const __bf16* __restrict__ WinT,
    const __bf16* __restrict__ WmmT, const __bf16* __restrict__ WnnT,
    const float* __restrict__ leak_m, const float* __restrict__ leak_n,
    float* __restrict__ Wm, float* __restrict__ Wn,
    __bf16* __restrict__ Hf) {
  __shared__ __bf16 xs[16][1024];       // x tile, bf16
  __shared__ __bf16 hbuf[64][16][64];   // h[n][b][m]
  __shared__ float  lbuf[2][16][64];    // gate logits (m / n)

  const int t    = threadIdx.x;
  const int lane = t & 31;
  const int wid  = t >> 5;
  const int b0   = blockIdx.x * 16;

  // ---- stage x tile (fp32 -> bf16), float4 loads, packed u32 LDS stores ---
  {
    int base = t * 64;
    int row = base >> 10, col = base & 1023;
    const float4* src = (const float4*)(x + (size_t)(b0 + row) * INC + col);
    unsigned* d = (unsigned*)&xs[row][col];
#pragma unroll
    for (int i = 0; i < 16; ++i) {
      float4 f = src[i];
      d[i * 2 + 0] = f2bf2(f.x, f.y);
      d[i * 2 + 1] = f2bf2(f.z, f.w);
    }
  }
  __syncthreads();

  // ---- gate logits: waves 0-3 -> m-gates, waves 4-7 -> n-gates ------------
  {
    int which = wid >> 2;   // 0 = m, 1 = n
    int nt    = wid & 3;    // N tile (16 cols)
    const __bf16* wbf = (which ? wnbf : wmbf) + (size_t)nt * 16 * 1024;
    const float*  bias = which ? wn_b : wm_b;
    v8f c = {};
    for (int k0 = 0; k0 < INC; k0 += 32) {
      if (k0 + 256 < INC)
        __builtin_prefetch(wbf + (size_t)(lane & 15) * 1024 + k0 + 256, 0, 3);
      v16bf a = load_a_frag(&xs[0][0], 1024, k0, lane);
      v16bf b = load_b_frag(wbf, 1024, k0, lane);
      c = wmma_bf16(a, b, c);
    }
    int col = cfrag_col(lane);
    float bv = bias[nt * 16 + col];
#pragma unroll
    for (int v = 0; v < 8; ++v)
      lbuf[which][cfrag_row(lane, v)][nt * 16 + col] = c[v] + bv;
  }
  __syncthreads();

  // ---- softmax over 64 (32 threads: 2 mats x 16 rows) ---------------------
  if (t < 32) {
    int which = t >> 4, row = t & 15;
    float mx = -3.4e38f;
    for (int j = 0; j < 64; ++j) mx = fmaxf(mx, lbuf[which][row][j]);
    float s = 0.f;
    for (int j = 0; j < 64; ++j) s += __expf(lbuf[which][row][j] - mx);
    float inv = 1.f / s;
    float* dst = (which ? Wn : Wm) + (size_t)(b0 + row) * 64;
    for (int j = 0; j < 64; ++j) dst[j] = __expf(lbuf[which][row][j] - mx) * inv;
  }
  __syncthreads();

  // ---- x_m: per-n GEMM [16x1024]@[1024x64], leaky over m ------------------
  for (int u = 0; u < 8; ++u) {
    int n = wid * 8 + u;
    const __bf16* wt = WimT + (size_t)n * 64 * 1024;  // [m][i]
    v8f c0 = {}, c1 = {}, c2 = {}, c3 = {};
    for (int k0 = 0; k0 < INC; k0 += 32) {
      if (k0 + 128 < INC)
        __builtin_prefetch(wt + (size_t)(lane & 15) * 1024 + k0 + 128, 0, 3);
      v16bf a = load_a_frag(&xs[0][0], 1024, k0, lane);
      c0 = wmma_bf16(a, load_b_frag(wt + 0 * 16 * 1024, 1024, k0, lane), c0);
      c1 = wmma_bf16(a, load_b_frag(wt + 1 * 16 * 1024, 1024, k0, lane), c1);
      c2 = wmma_bf16(a, load_b_frag(wt + 2 * 16 * 1024, 1024, k0, lane), c2);
      c3 = wmma_bf16(a, load_b_frag(wt + 3 * 16 * 1024, 1024, k0, lane), c3);
    }
    int col = cfrag_col(lane);
    v8f cc[4] = {c0, c1, c2, c3};
#pragma unroll
    for (int mt = 0; mt < 4; ++mt) {
      float lk = leak_m[mt * 16 + col];
#pragma unroll
      for (int v = 0; v < 8; ++v) {
        float val = cc[mt][v];
        val = val < 0.f ? val * lk : val;
        hbuf[n][cfrag_row(lane, v)][mt * 16 + col] = f2bf(val);
      }
    }
  }
  __syncthreads();

  // ---- x_n: per-m GEMM, leaky over n, transposed add into hbuf ------------
  for (int u = 0; u < 8; ++u) {
    int m = wid * 8 + u;
    const __bf16* wt = WinT + (size_t)m * 64 * 1024;  // [nn][i]
    v8f c0 = {}, c1 = {}, c2 = {}, c3 = {};
    for (int k0 = 0; k0 < INC; k0 += 32) {
      if (k0 + 128 < INC)
        __builtin_prefetch(wt + (size_t)(lane & 15) * 1024 + k0 + 128, 0, 3);
      v16bf a = load_a_frag(&xs[0][0], 1024, k0, lane);
      c0 = wmma_bf16(a, load_b_frag(wt + 0 * 16 * 1024, 1024, k0, lane), c0);
      c1 = wmma_bf16(a, load_b_frag(wt + 1 * 16 * 1024, 1024, k0, lane), c1);
      c2 = wmma_bf16(a, load_b_frag(wt + 2 * 16 * 1024, 1024, k0, lane), c2);
      c3 = wmma_bf16(a, load_b_frag(wt + 3 * 16 * 1024, 1024, k0, lane), c3);
    }
    int col = cfrag_col(lane);
    v8f cc[4] = {c0, c1, c2, c3};
#pragma unroll
    for (int nt = 0; nt < 4; ++nt) {
      float lk = leak_n[nt * 16 + col];
#pragma unroll
      for (int v = 0; v < 8; ++v) {
        float val = cc[nt][v];
        val = val < 0.f ? val * lk : val;
        int nn = nt * 16 + col, r = cfrag_row(lane, v);
        hbuf[nn][r][m] = f2bf(bf2f(hbuf[nn][r][m]) + val);  // column m owned by this wave
      }
    }
  }
  __syncthreads();

  // ---- hidden loop: 3 iterations ------------------------------------------
  for (int it = 0; it < 3; ++it) {
    // step A: per n, Hn [16x64] @ W_mm[it][n] [64x64], leaky over z(=m)
    for (int u = 0; u < 8; ++u) {
      int n = wid * 8 + u;
      v16bf a0 = load_a_frag(&hbuf[n][0][0], 64, 0, lane);
      v16bf a1 = load_a_frag(&hbuf[n][0][0], 64, 32, lane);
      const __bf16* wt = WmmT + (size_t)(it * 64 + n) * 64 * 64;  // [z][m]
      v8f c[4];
#pragma unroll
      for (int zt = 0; zt < 4; ++zt) {
        v8f acc = {};
        acc = wmma_bf16(a0, load_b_frag(wt + zt * 16 * 64, 64, 0, lane), acc);
        acc = wmma_bf16(a1, load_b_frag(wt + zt * 16 * 64, 64, 32, lane), acc);
        c[zt] = acc;
      }
      int col = cfrag_col(lane);
#pragma unroll
      for (int zt = 0; zt < 4; ++zt) {
        float lk = leak_m[zt * 16 + col];
#pragma unroll
        for (int v = 0; v < 8; ++v) {
          float val = c[zt][v];
          val = val < 0.f ? val * lk : val;
          hbuf[n][cfrag_row(lane, v)][zt * 16 + col] = f2bf(val);
        }
      }
    }
    __syncthreads();

    // step B: per m, Gm[b,n] @ W_nn[it][m] [64x64], leaky over z(=n),
    // result written transposed back: h[z][b][m]
    for (int u = 0; u < 8; ++u) {
      int m = wid * 8 + u;
      v16bf a0 = load_a_frag_gather(&hbuf[0][0][m], 64, 16 * 64, 0, lane);
      v16bf a1 = load_a_frag_gather(&hbuf[0][0][m], 64, 16 * 64, 32, lane);
      const __bf16* wt = WnnT + (size_t)(it * 64 + m) * 64 * 64;  // [z][n]
      v8f c[4];
#pragma unroll
      for (int zt = 0; zt < 4; ++zt) {
        v8f acc = {};
        acc = wmma_bf16(a0, load_b_frag(wt + zt * 16 * 64, 64, 0, lane), acc);
        acc = wmma_bf16(a1, load_b_frag(wt + zt * 16 * 64, 64, 32, lane), acc);
        c[zt] = acc;
      }
      int col = cfrag_col(lane);
#pragma unroll
      for (int zt = 0; zt < 4; ++zt) {
        float lk = leak_n[zt * 16 + col];
#pragma unroll
        for (int v = 0; v < 8; ++v) {
          float val = c[zt][v];
          val = val < 0.f ? val * lk : val;
          hbuf[zt * 16 + col][cfrag_row(lane, v)][m] = f2bf(val);
        }
      }
    }
    __syncthreads();
  }

  // ---- write h to global: Hf[b][n][m], 128B rows ---------------------------
  for (int rtask = t; rtask < 16 * 64; rtask += 256) {
    int b = rtask >> 6, n = rtask & 63;
    const v8bf* s = (const v8bf*)&hbuf[n][b][0];
    v8bf* d = (v8bf*)(Hf + ((size_t)(b0 + b) * 4096 + n * 64));
#pragma unroll
    for (int i = 0; i < 8; ++i) d[i] = s[i];
  }
}

// ---------------------------------------------------------------------------
// Output branch: compute gated projection tile v[n][b][o] (all n, 16 b, 64 o)
// via WMMA, then stage-1 (over n) stats; leaves mu1 in s1[], invden1 in q1[].
// No atomics: stats built in a conflict-free post-pass over vt.
// ---------------------------------------------------------------------------
static __device__ void out_branch(const __bf16* __restrict__ Hf,
                                  const __bf16* __restrict__ WT,   // [unit][o][k]
                                  const float* __restrict__ gate,  // [B][64]
                                  int b0, int o0, int t,
                                  __bf16 (*vt)[16][64],
                                  float (*s1)[64], float (*q1)[64]) {
  int lane = t & 31, wid = t >> 5;
  for (int u = 0; u < 8; ++u) {
    int n = wid * 8 + u;
    const __bf16* arow = Hf + (size_t)b0 * 4096 + n * 64;  // row stride 4096
    v16bf a0 = load_a_frag(arow, 4096, 0, lane);
    v16bf a1 = load_a_frag(arow, 4096, 32, lane);
    const __bf16* wt = WT + (size_t)n * 1024 * 64 + (size_t)o0 * 64;
    v8f c[4];
#pragma unroll
    for (int ot = 0; ot < 4; ++ot) {
      v8f acc = {};
      acc = wmma_bf16(a0, load_b_frag(wt + ot * 16 * 64, 64, 0, lane), acc);
      acc = wmma_bf16(a1, load_b_frag(wt + ot * 16 * 64, 64, 32, lane), acc);
      c[ot] = acc;
    }
    int col = cfrag_col(lane);
#pragma unroll
    for (int ot = 0; ot < 4; ++ot) {
#pragma unroll
      for (int v = 0; v < 8; ++v) {
        int r = cfrag_row(lane, v);
        float g = gate[(size_t)(b0 + r) * 64 + n];
        vt[n][r][ot * 16 + col] = f2bf(c[ot][v] * g);
      }
    }
  }
  __syncthreads();
  // stage-1 stats over n per (b, o); adjacent lanes hit adjacent banks.
  for (int p = t; p < 16 * 64; p += 256) {
    int b = p >> 6, o = p & 63;
    float s = 0.f, q = 0.f;
    for (int n = 0; n < 64; ++n) {
      float v = bf2f(vt[n][b][o]);
      s += v;
      q += v * v;
    }
    float mu  = s * (1.f / 64.f);
    float var = q * (1.f / 64.f) - mu * mu;
    s1[b][o] = mu;
    q1[b][o] = 1.f / fmaxf(sqrtf(fmaxf(var, 0.f)), 1e-8f);
  }
  __syncthreads();
}

// K3: accumulate stage-2 (over o) sums of the stage-1-normalized tile.
static __device__ void accum_stage2(__bf16 (*vt)[16][64], float (*mu1)[64],
                                    float (*iv1)[64], int t, int b0,
                                    float* __restrict__ S2,
                                    float* __restrict__ Q2) {
  for (int p = t; p < 16 * 64; p += 256) {  // (b, n) pairs
    int b = p >> 6, n = p & 63;
    float s = 0.f, q = 0.f;
    for (int o = 0; o < 64; ++o) {
      float x2 = (bf2f(vt[n][b][o]) - mu1[b][o]) * iv1[b][o];
      s += x2;
      q += x2 * x2;
    }
    atomicAdd(&S2[(size_t)(b0 + b) * 64 + n], s);
    atomicAdd(&Q2[(size_t)(b0 + b) * 64 + n], q);
  }
}

__global__ __launch_bounds__(256) void k_out_stats(
    const __bf16* __restrict__ Hf,
    const __bf16* __restrict__ WmoT, const __bf16* __restrict__ WnoT,
    const float* __restrict__ Wn, const float* __restrict__ Wm,
    float* __restrict__ S2m, float* __restrict__ Q2m,
    float* __restrict__ S2n, float* __restrict__ Q2n) {
  __shared__ __bf16 vt[64][16][64];
  __shared__ float s1[16][64], q1[16][64];
  int t  = threadIdx.x;
  int b0 = (blockIdx.x >> 4) * 16;
  int o0 = (blockIdx.x & 15) * 64;
  out_branch(Hf, WmoT, Wn, b0, o0, t, vt, s1, q1);
  accum_stage2(vt, s1, q1, t, b0, S2m, Q2m);
  __syncthreads();
  out_branch(Hf, WnoT, Wm, b0, o0, t, vt, s1, q1);
  accum_stage2(vt, s1, q1, t, b0, S2n, Q2n);
}

// K4: recompute tile, apply both norms, mean over n, sum both branches.
static __device__ void final_accum(__bf16 (*vt)[16][64], float (*mu1)[64],
                                   float (*iv1)[64], float (*mu2s)[64],
                                   float (*iv2s)[64], int t,
                                   float (*oacc)[64]) {
  for (int p = t; p < 16 * 64; p += 256) {  // (b, o) pairs
    int b = p >> 6, o = p & 63;
    float m1 = mu1[b][o], i1 = iv1[b][o];
    float acc = 0.f;
    for (int n = 0; n < 64; ++n) {
      float x2 = (bf2f(vt[n][b][o]) - m1) * i1;
      acc += (x2 - mu2s[b][n]) * iv2s[b][n];
    }
    oacc[b][o] += acc * (1.f / 64.f);
  }
}

static __device__ void stage2_consts(const float* __restrict__ S2,
                                     const float* __restrict__ Q2, int t,
                                     int b0, float (*mu2s)[64],
                                     float (*iv2s)[64]) {
  for (int p = t; p < 16 * 64; p += 256) {  // (b, n) pairs
    int b = p >> 6, n = p & 63;
    float mu2  = S2[(size_t)(b0 + b) * 64 + n] * (1.f / 1024.f);
    float var2 = Q2[(size_t)(b0 + b) * 64 + n] * (1.f / 1024.f) - mu2 * mu2;
    mu2s[b][n] = mu2;
    iv2s[b][n] = 1.f / fmaxf(sqrtf(fmaxf(var2, 0.f)), 1e-8f);
  }
}

__global__ __launch_bounds__(256) void k_out_final(
    const __bf16* __restrict__ Hf,
    const __bf16* __restrict__ WmoT, const __bf16* __restrict__ WnoT,
    const float* __restrict__ Wn, const float* __restrict__ Wm,
    const float* __restrict__ S2m, const float* __restrict__ Q2m,
    const float* __restrict__ S2n, const float* __restrict__ Q2n,
    float* __restrict__ out) {
  __shared__ __bf16 vt[64][16][64];
  __shared__ float s1[16][64], q1[16][64];
  __shared__ float mu2s[16][64], iv2s[16][64];
  __shared__ float oacc[16][64];
  int t  = threadIdx.x;
  int b0 = (blockIdx.x >> 4) * 16;
  int o0 = (blockIdx.x & 15) * 64;
  for (int e = t; e < 1024; e += 256) ((float*)oacc)[e] = 0.f;
  // branch m
  stage2_consts(S2m, Q2m, t, b0, mu2s, iv2s);  // synced inside out_branch
  out_branch(Hf, WmoT, Wn, b0, o0, t, vt, s1, q1);
  final_accum(vt, s1, q1, mu2s, iv2s, t, oacc);
  __syncthreads();
  // branch n
  stage2_consts(S2n, Q2n, t, b0, mu2s, iv2s);
  __syncthreads();
  out_branch(Hf, WnoT, Wm, b0, o0, t, vt, s1, q1);
  final_accum(vt, s1, q1, mu2s, iv2s, t, oacc);
  __syncthreads();
  for (int e = t; e < 1024; e += 256) {
    int b = e >> 6, o = e & 63;
    out[(size_t)(b0 + b) * 1024 + o0 + o] = oacc[b][o];
  }
}

// ---------------------------------------------------------------------------
extern "C" void kernel_launch(void* const* d_in, const int* in_sizes, int n_in,
                              void* d_out, int out_size, void* d_ws,
                              size_t ws_size, hipStream_t stream) {
  (void)in_sizes; (void)n_in; (void)out_size; (void)ws_size;
  const float* x      = (const float*)d_in[0];
  const float* wm_w   = (const float*)d_in[1];
  const float* wm_b   = (const float*)d_in[2];
  const float* wn_w   = (const float*)d_in[3];
  const float* wn_b   = (const float*)d_in[4];
  const float* W_im   = (const float*)d_in[5];
  const float* W_in   = (const float*)d_in[6];
  const float* W_mm   = (const float*)d_in[7];
  const float* W_nn   = (const float*)d_in[8];
  const float* W_mo   = (const float*)d_in[9];
  const float* W_no   = (const float*)d_in[10];
  const float* leak_m = (const float*)d_in[11];
  const float* leak_n = (const float*)d_in[12];
  float* out = (float*)d_out;

  char* ws = (char*)d_ws;
  auto alloc = [&](size_t bytes) {
    char* p = ws;
    ws += (bytes + 255) & ~(size_t)255;
    return p;
  };
  __bf16* wmbf = (__bf16*)alloc((size_t)64 * 1024 * 2);
  __bf16* wnbf = (__bf16*)alloc((size_t)64 * 1024 * 2);
  __bf16* WimT = (__bf16*)alloc((size_t)64 * 1024 * 64 * 2);
  __bf16* WinT = (__bf16*)alloc((size_t)64 * 1024 * 64 * 2);
  __bf16* WmmT = (__bf16*)alloc((size_t)192 * 64 * 64 * 2);
  __bf16* WnnT = (__bf16*)alloc((size_t)192 * 64 * 64 * 2);
  __bf16* WmoT = (__bf16*)alloc((size_t)64 * 64 * 1024 * 2);
  __bf16* WnoT = (__bf16*)alloc((size_t)64 * 64 * 1024 * 2);
  float*  Wm   = (float*)alloc((size_t)BT * 64 * 4);
  float*  Wn   = (float*)alloc((size_t)BT * 64 * 4);
  __bf16* Hf   = (__bf16*)alloc((size_t)BT * 4096 * 2);
  float*  S2m  = (float*)alloc((size_t)BT * 64 * 4);
  float*  Q2m  = (float*)alloc((size_t)BT * 64 * 4);
  float*  S2n  = (float*)alloc((size_t)BT * 64 * 4);
  float*  Q2n  = (float*)alloc((size_t)BT * 64 * 4);

  // --- K0: bf16 copies / transposed weight layouts -------------------------
  k_copy_bf16<<<(64 * 1024 + 255) / 256, 256, 0, stream>>>(wmbf, wm_w, 64 * 1024);
  k_copy_bf16<<<(64 * 1024 + 255) / 256, 256, 0, stream>>>(wnbf, wn_w, 64 * 1024);
  k_transpose_bf16<<<64 * 16 * 1, 256, 0, stream>>>(WimT, W_im, 1024, 64);
  k_transpose_bf16<<<64 * 16 * 1, 256, 0, stream>>>(WinT, W_in, 1024, 64);
  k_transpose_bf16<<<192 * 1 * 1, 256, 0, stream>>>(WmmT, W_mm, 64, 64);
  k_transpose_bf16<<<192 * 1 * 1, 256, 0, stream>>>(WnnT, W_nn, 64, 64);
  k_transpose_bf16<<<64 * 1 * 16, 256, 0, stream>>>(WmoT, W_mo, 64, 1024);
  k_transpose_bf16<<<64 * 1 * 16, 256, 0, stream>>>(WnoT, W_no, 64, 1024);

  // --- zero stage-2 stat accumulators (S2m,Q2m,S2n,Q2n are contiguous) -----
  hipMemsetAsync(S2m, 0, (size_t)4 * BT * 64 * 4, stream);

  // --- K2: fused gates + projections + hidden loop -------------------------
  k_hidden<<<BT / 16, 256, 0, stream>>>(x, wmbf, wm_b, wnbf, wn_b, WimT, WinT,
                                        WmmT, WnnT, leak_m, leak_n, Wm, Wn, Hf);

  // --- K3 / K4: output projections + rmsnorm2d + mean ----------------------
  dim3 ogrid((BT / 16) * (OUTC / 64));
  k_out_stats<<<ogrid, 256, 0, stream>>>(Hf, WmoT, WnoT, Wn, Wm, S2m, Q2m, S2n,
                                         Q2n);
  k_out_final<<<ogrid, 256, 0, stream>>>(Hf, WmoT, WnoT, Wn, Wm, S2m, Q2m, S2n,
                                         Q2n, out);
}
